// CrossModalAttention3D_47777216200966
// MI455X (gfx1250) — matrix operand order repaired and checked by reading, not compiled, not run on hardware
//
#include <hip/hip_runtime.h>
#include <hip/hip_bf16.h>
#include <math.h>

#define B_  2
#define C_  128
#define CH_ 64
#define N_  4096          // 16*16*16
#define BN_EPS 1e-5f

typedef __attribute__((ext_vector_type(16))) _Float16 v16h;
typedef __attribute__((ext_vector_type(8)))  _Float16 v8h;
typedef __attribute__((ext_vector_type(8)))  float    v8f;

#define WMMA_F16(Af, Bf, Cf) \
  __builtin_amdgcn_wmma_f32_16x16x32_f16(false, (Af), false, (Bf), (short)0, (Cf), false, false)

#define SHUF16(lo, hh) __builtin_shufflevector((lo), (hh), 0,1,2,3,4,5,6,7,8,9,10,11,12,13,14,15)

// ---------------- kernel 1: pooled = mean(mri+pet) over spatial ----------------
__global__ void k_pool(const float* __restrict__ mri, const float* __restrict__ pet,
                       float* __restrict__ pooled) {
  int bc = blockIdx.x;                       // b*C + c
  const float* m = mri + (size_t)bc * N_;
  const float* p = pet + (size_t)bc * N_;
  float s = 0.f;
  for (int i = threadIdx.x; i < N_; i += 256) s += m[i] + p[i];
  __shared__ float red[256];
  red[threadIdx.x] = s; __syncthreads();
  for (int st = 128; st > 0; st >>= 1) {
    if ((int)threadIdx.x < st) red[threadIdx.x] += red[threadIdx.x + st];
    __syncthreads();
  }
  if (threadIdx.x == 0) pooled[bc] = red[0] * (1.f / (float)N_);
}

// ---------------- kernel 2: squeeze-excite MLP -> cw ----------------
__global__ void k_mlp(const float* __restrict__ pooled,
                      const float* __restrict__ w1, const float* __restrict__ b1,
                      const float* __restrict__ w2, const float* __restrict__ b2,
                      float* __restrict__ cw) {
  __shared__ float ts[B_][CH_];
  int o = threadIdx.x;                       // 0..127
  if (o < CH_) {
    for (int b = 0; b < B_; ++b) {
      float a = b1[o];
      for (int c = 0; c < C_; ++c) a = fmaf(w1[o*C_ + c], pooled[b*C_ + c], a);
      ts[b][o] = fmaxf(a, 0.f);
    }
  }
  __syncthreads();
  for (int b = 0; b < B_; ++b) {
    float a = b2[o];
    for (int h = 0; h < CH_; ++h) a = fmaf(w2[o*CH_ + h], ts[b][h], a);
    cw[b*C_ + o] = 1.f / (1.f + __expf(-a));
  }
}

// ------- kernel 3: Q/K/V 1x1 convs with cw folded in; f16 WMMA-friendly layouts -------
// Qh,Kh: [b][p][c] f16 ;  Vth: [b][c][p] f16
__global__ void k_qkv(const float* __restrict__ mri, const float* __restrict__ pet,
                      const float* __restrict__ cw,
                      const float* __restrict__ qw, const float* __restrict__ qb,
                      const float* __restrict__ kw, const float* __restrict__ kb,
                      const float* __restrict__ vw, const float* __restrict__ vb,
                      _Float16* __restrict__ Qh, _Float16* __restrict__ Kh,
                      _Float16* __restrict__ Vth) {
  int tid = blockIdx.x * 256 + threadIdx.x;  // b*N*C total
  int o = tid & (C_ - 1);
  int p = (tid >> 7) & (N_ - 1);
  int b = tid >> 19;
  const float* cwb = cw + b * C_;
  const float* mb  = mri + (size_t)b * C_ * N_ + p;
  const float* pb  = pet + (size_t)b * C_ * N_ + p;
  float aq = qb[o], ak = kb[o], av = vb[o];
  for (int c = 0; c < C_; ++c) {
    float s  = cwb[c];
    float mw = mb[(size_t)c * N_] * s;
    float pw = pb[(size_t)c * N_] * s;
    aq = fmaf(qw[o*C_ + c], mw, aq);
    ak = fmaf(kw[o*C_ + c], pw, ak);
    av = fmaf(vw[o*C_ + c], pw, av);
  }
  size_t rc = ((size_t)b * N_ + p) * C_ + o;
  Qh[rc] = (_Float16)aq;
  Kh[rc] = (_Float16)ak;
  Vth[((size_t)b * C_ + o) * N_ + p] = (_Float16)av;
}

// ---------------- kernel 4: flash attention, one wave per 16-query tile ----------------
// 64-key chunks: softmax bookkeeping (shuffles, rescale, exp) amortized over 2x the
// WMMA work vs a 32-key chunk. x[b][c][p] = softmax(Q^T K)[p,:] @ V^T + mri*cw
__launch_bounds__(128, 1)
__global__ void k_attn(const _Float16* __restrict__ Qh, const _Float16* __restrict__ Kh,
                       const _Float16* __restrict__ Vth,
                       const float* __restrict__ mri, const float* __restrict__ cw,
                       float* __restrict__ x) {
  __shared__ _Float16 pbuf[4][16 * 64];      // per-wave C-layout -> A-layout bounce (2KB/wave)
  const int lane = threadIdx.x & 31;
  const int wid  = threadIdx.x >> 5;
  const int wgl  = blockIdx.x * 4 + wid;     // 0..511
  const int b    = wgl >> 8;
  const int q0   = (wgl & 255) << 4;
  const int nn   = lane & 15;
  const int hi   = lane >> 4;

  const _Float16* Qb = Qh  + (size_t)b * N_ * C_;
  const _Float16* Kb = Kh  + (size_t)b * N_ * C_;
  const _Float16* Vb = Vth + (size_t)b * C_ * N_;

  // Q tile, A-layout, K=C=128 -> 4 ksteps of 32 (kept in registers whole kernel)
  v16h aq[4];
  #pragma unroll
  for (int ks = 0; ks < 4; ++ks) {
    const _Float16* r = Qb + (size_t)(q0 + nn) * C_ + ks * 32;
    v8h lo = *(const v8h*)(r + hi * 16);
    v8h hh = *(const v8h*)(r + hi * 16 + 8);
    aq[ks] = SHUF16(lo, hh);
  }

  v8f zero = {};
  v8f acc[8];                                 // 16q x 128c f32 tile, rows m = r + 8*hi
  #pragma unroll
  for (int t = 0; t < 8; ++t) acc[t] = zero;
  float rmax[8], rsum[8];
  #pragma unroll
  for (int r = 0; r < 8; ++r) { rmax[r] = -__builtin_inff(); rsum[r] = 0.f; }

  _Float16* pb = pbuf[wid];

  for (int P0 = 0; P0 < N_; P0 += 64) {
    // ---- scores chunk: four 16x16 key tiles, K=128 (16 WMMAs) ----
    v8f sfr[4];
    #pragma unroll
    for (int h2 = 0; h2 < 4; ++h2) {
      v8f sacc = zero;
      const _Float16* krow = Kb + (size_t)(P0 + h2 * 16 + nn) * C_ + hi * 16;
      #pragma unroll
      for (int ks = 0; ks < 4; ++ks) {
        v16h bk = *(const v16h*)(krow + ks * 32);
        sacc = WMMA_F16(aq[ks], bk, sacc);
      }
      sfr[h2] = sacc;
    }
    // ---- online softmax update (per-lane row stats: row m = r + 8*hi) ----
    #pragma unroll
    for (int r = 0; r < 8; ++r) {
      float m = fmaxf(fmaxf(sfr[0][r], sfr[1][r]), fmaxf(sfr[2][r], sfr[3][r]));
      m = fmaxf(m, __shfl_xor(m, 1, 16));
      m = fmaxf(m, __shfl_xor(m, 2, 16));
      m = fmaxf(m, __shfl_xor(m, 4, 16));
      m = fmaxf(m, __shfl_xor(m, 8, 16));
      float nm = fmaxf(rmax[r], m);
      float f  = __expf(rmax[r] - nm);
      rmax[r]  = nm;
      float pv[4];
      float ps = 0.f;
      #pragma unroll
      for (int h2 = 0; h2 < 4; ++h2) {
        pv[h2] = __expf(sfr[h2][r] - nm);
        ps += pv[h2];
      }
      ps += __shfl_xor(ps, 1, 16);
      ps += __shfl_xor(ps, 2, 16);
      ps += __shfl_xor(ps, 4, 16);
      ps += __shfl_xor(ps, 8, 16);
      rsum[r] = rsum[r] * f + ps;
      #pragma unroll
      for (int t = 0; t < 8; ++t) acc[t][r] *= f;
      int row = r + 8 * hi;
      #pragma unroll
      for (int h2 = 0; h2 < 4; ++h2)
        pb[row * 64 + h2 * 16 + nn] = (_Float16)pv[h2];   // k = h2*16 + nn
    }
    // ---- P tile back as A-layout (2 ksteps), acc += P @ V^T over 8 c-tiles (16 WMMAs) ----
    #pragma unroll
    for (int ks2 = 0; ks2 < 2; ++ks2) {
      const _Float16* ar = pb + nn * 64 + ks2 * 32 + hi * 16;
      v8h lo = *(const v8h*)(ar);
      v8h hh = *(const v8h*)(ar + 8);
      v16h ap = SHUF16(lo, hh);
      #pragma unroll
      for (int t = 0; t < 8; ++t) {
        const _Float16* vrow = Vb + (size_t)(t * 16 + nn) * N_ + P0 + ks2 * 32 + hi * 16;
        v16h bv = *(const v16h*)vrow;
        acc[t] = WMMA_F16(ap, bv, acc[t]);
      }
    }
  }

  // ---- epilogue: normalize + residual mri*cw, write x[b][c][p] ----
  const float* mrb = mri + (size_t)b * C_ * N_;
  float* xb = x + (size_t)b * C_ * N_;
  #pragma unroll
  for (int t = 0; t < 8; ++t) {
    int c = t * 16 + nn;
    float scale = cw[b * C_ + c];
    size_t base = (size_t)c * N_ + q0 + 8 * hi;
    #pragma unroll
    for (int r = 0; r < 8; ++r)
      xb[base + r] = acc[t][r] / rsum[r] + mrb[base + r] * scale;
  }
}

// ---------------- kernel 5: 3x3x3 conv + ReLU + BN -> yh [b][p][co] f16 ----------------
__global__ void k_conv(const float* __restrict__ x, const float* __restrict__ w,
                       const float* __restrict__ bias,
                       const float* __restrict__ gamma, const float* __restrict__ beta,
                       const float* __restrict__ mean,  const float* __restrict__ var,
                       _Float16* __restrict__ yh) {
  int blk = blockIdx.x;                      // b*CH*16 + co*16 + d
  int d  = blk & 15;
  int co = (blk >> 4) & 63;
  int b  = blk >> 10;
  int h  = threadIdx.x >> 4;
  int wv = threadIdx.x & 15;
  const float* xb = x + (size_t)b * C_ * N_;
  const float* wc = w + (size_t)co * C_ * 27;
  float acc = bias[co];
  for (int ci = 0; ci < C_; ++ci) {
    const float* xc = xb + (size_t)ci * N_;
    const float* wk = wc + ci * 27;
    #pragma unroll
    for (int kd = 0; kd < 3; ++kd) {
      int zd = d + kd - 1;
      if ((unsigned)zd >= 16u) continue;
      #pragma unroll
      for (int kh = 0; kh < 3; ++kh) {
        int zh = h + kh - 1;
        if ((unsigned)zh >= 16u) continue;
        #pragma unroll
        for (int kw = 0; kw < 3; ++kw) {
          int zw = wv + kw - 1;
          if ((unsigned)zw >= 16u) continue;
          acc = fmaf(wk[kd*9 + kh*3 + kw], xc[zd*256 + zh*16 + zw], acc);
        }
      }
    }
  }
  acc = fmaxf(acc, 0.f);
  acc = (acc - mean[co]) * (gamma[co] * rsqrtf(var[co] + BN_EPS)) + beta[co];
  int p = d * 256 + h * 16 + wv;
  yh[((size_t)b * N_ + p) * CH_ + co] = (_Float16)acc;
}

// ---------------- helper: f32 -> f16 weight convert ----------------
__global__ void k_cvtw(const float* __restrict__ w, _Float16* __restrict__ wh, int n) {
  int i = blockIdx.x * 256 + threadIdx.x;
  if (i < n) wh[i] = (_Float16)w[i];
}

// ---------------- kernel 6: final 1x1 conv (64->128) as WMMA GEMM ----------------
__launch_bounds__(256, 1)
__global__ void k_out(const _Float16* __restrict__ yh, const _Float16* __restrict__ c2wh,
                      const float* __restrict__ c2b, float* __restrict__ out) {
  int lane = threadIdx.x & 31;
  int wid  = threadIdx.x >> 5;               // o-tile 0..7
  int blk  = blockIdx.x;                     // b*256 + ptile
  int b    = blk >> 8;
  int p0   = (blk & 255) << 4;
  int nn   = lane & 15, hi = lane >> 4;
  int o0   = wid * 16;
  v8f acc = {};
  #pragma unroll
  for (int ks = 0; ks < 2; ++ks) {
    const _Float16* ar = yh + ((size_t)b * N_ + p0 + nn) * CH_ + ks * 32;
    v8h lo = *(const v8h*)(ar + hi * 16);
    v8h hh = *(const v8h*)(ar + hi * 16 + 8);
    v16h a  = SHUF16(lo, hh);
    v16h bm = *(const v16h*)(c2wh + (size_t)(o0 + nn) * CH_ + ks * 32 + hi * 16);
    acc = WMMA_F16(a, bm, acc);
  }
  int o = o0 + nn;
  float bias = c2b[o];
  float* ob = out + ((size_t)b * C_ + o) * N_ + p0 + 8 * hi;
  #pragma unroll
  for (int r = 0; r < 8; ++r) ob[r] = acc[r] + bias;
}

extern "C" void kernel_launch(void* const* d_in, const int* in_sizes, int n_in,
                              void* d_out, int out_size, void* d_ws, size_t ws_size,
                              hipStream_t stream) {
  const float* mri   = (const float*)d_in[0];
  const float* pet   = (const float*)d_in[1];
  const float* qw    = (const float*)d_in[2];
  const float* qb    = (const float*)d_in[3];
  const float* kw    = (const float*)d_in[4];
  const float* kb    = (const float*)d_in[5];
  const float* vw    = (const float*)d_in[6];
  const float* vb    = (const float*)d_in[7];
  const float* ca_w1 = (const float*)d_in[8];
  const float* ca_b1 = (const float*)d_in[9];
  const float* ca_w2 = (const float*)d_in[10];
  const float* ca_b2 = (const float*)d_in[11];
  const float* c1_w  = (const float*)d_in[12];
  const float* c1_b  = (const float*)d_in[13];
  const float* bn_g  = (const float*)d_in[14];
  const float* bn_b  = (const float*)d_in[15];
  const float* bn_m  = (const float*)d_in[16];
  const float* bn_v  = (const float*)d_in[17];
  const float* c2_w  = (const float*)d_in[18];
  const float* c2_b  = (const float*)d_in[19];
  float* out = (float*)d_out;

  char* w = (char*)d_ws;
  float*    pooled = (float*)(w + 0);                       // 1 KB
  float*    cwv    = (float*)(w + 1024);                    // 1 KB
  _Float16* c2wh   = (_Float16*)(w + 2048);                 // 16 KB
  size_t off = 32768;
  _Float16* Qh  = (_Float16*)(w + off); off += (size_t)B_ * N_ * C_ * 2;   // 2 MB
  _Float16* Kh  = (_Float16*)(w + off); off += (size_t)B_ * N_ * C_ * 2;   // 2 MB
  _Float16* Vth = (_Float16*)(w + off); off += (size_t)B_ * C_ * N_ * 2;   // 2 MB
  float*    x   = (float*)(w + off);    off += (size_t)B_ * C_ * N_ * 4;   // 4 MB
  _Float16* yh  = (_Float16*)(w + off);                                     // 1 MB

  k_pool<<<B_ * C_, 256, 0, stream>>>(mri, pet, pooled);
  k_mlp<<<1, 128, 0, stream>>>(pooled, ca_w1, ca_b1, ca_w2, ca_b2, cwv);
  k_qkv<<<(B_ * N_ * C_) / 256, 256, 0, stream>>>(mri, pet, cwv, qw, qb, kw, kb, vw, vb,
                                                  Qh, Kh, Vth);
  k_cvtw<<<(C_ * CH_ + 255) / 256, 256, 0, stream>>>(c2_w, c2wh, C_ * CH_);
  k_attn<<<(B_ * (N_ / 16)) / 4, 128, 0, stream>>>(Qh, Kh, Vth, mri, cwv, x);
  k_conv<<<B_ * CH_ * 16, 256, 0, stream>>>(x, c1_w, c1_b, bn_g, bn_b, bn_m, bn_v, yh);
  k_out<<<B_ * (N_ / 16), 256, 0, stream>>>(yh, c2wh, c2_b, out);
}
